// S2VT_26036091748958
// MI455X (gfx1250) — matrix operand order repaired and checked by reading, not compiled, not run on hardware
//
#include <hip/hip_runtime.h>
#include <hip/hip_bf16.h>

#define B_SZ   64
#define T_SEQ  80
#define F_DIM  4096
#define E_DIM  512
#define H_DIM  512
#define DW_DIM 512
#define V_DIM  13000
#define V_PAD  13056      // next multiple of 64
#define G4     2048       // 4*H
#define START_ID 1

typedef __bf16 bf16_t;
typedef bf16_t v16bf __attribute__((ext_vector_type(16)));
typedef bf16_t v2bf  __attribute__((ext_vector_type(2)));
typedef float  v8f   __attribute__((ext_vector_type(8)));

// float -> bf16 bits, RNE.  Native fptrunc (RNE by definition) lets the
// backend pick the HW convert.
__device__ __forceinline__ unsigned short f2bf(float x) {
  union { bf16_t b; unsigned short s; } c;
  c.b = (bf16_t)x;
  return c.s;
}
// packed f32x2 -> bf16x2; vector form lets the backend form a packed cvt
__device__ __forceinline__ unsigned int f2bf_pk(float x, float y) {
#if defined(__AMDGCN__) && __has_builtin(__builtin_amdgcn_cvt_pk_bf16_f32)
  union { v2bf v; unsigned int u; } c;
  c.v = __builtin_amdgcn_cvt_pk_bf16_f32(x, y);
  return c.u;
#else
  union { v2bf v; unsigned int u; } c;
  c.v[0] = (bf16_t)x;
  c.v[1] = (bf16_t)y;
  return c.u;
#endif
}

union V16U {
  v16bf v;
  unsigned short u[16];
  unsigned int w[8];
  uint4 q[2];
};

// ---------------------------------------------------------------------------
// Generic WMMA GEMM:  Out[M x Nreal] = A[M x K] * B^T (B stored N x K, bf16)
//   AF32     : A is fp32 (converted to bf16 in-register) else bf16 (ushort bits)
//   HAS_CIN  : add Cin[mrow*ldout + n] in epilogue (fp32, same layout as out)
//   HAS_BIAS : add bias[n]
//   OUT_BF16 : store bf16 bits (ushort) else fp32
//   SWAP     : out_row = (mrow % inner) * (M/inner) + mrow / inner
// One wave computes a 16x64 strip; Npad must be a multiple of 64, M of 16,
// K of 32.  Early-exit is wave-uniform so EXEC is all-ones around WMMA.
// ---------------------------------------------------------------------------
template<bool AF32, bool HAS_CIN, bool HAS_BIAS, bool OUT_BF16, bool SWAP>
__global__ void __launch_bounds__(128)
gemm_wmma(const void* __restrict__ Ap, const unsigned short* __restrict__ Bp,
          const float* __restrict__ Cin, const float* __restrict__ bias,
          void* __restrict__ Op,
          int M, int K, int Npad, int Nreal, int ldout, int inner)
{
  const int lane    = threadIdx.x & 31;
  const int wid     = (blockIdx.x * blockDim.x + threadIdx.x) >> 5;
  const int mtiles  = M >> 4;
  const int ngroups = Npad >> 6;
  if (wid >= mtiles * ngroups) return;
  const int mt = wid % mtiles;
  const int ng = wid / mtiles;
  const int m0 = mt << 4;
  const int n0 = ng << 6;
  const int lhalf = lane & 15;
  const int lsel  = lane >> 4;

  const int am = m0 + lhalf;        // A row this lane reads
  const int ak = lsel << 3;         // A K sub-offset: 0 or 8
  const int bk = lsel << 3;         // B K sub-offset: 0 or 8

  v8f acc[4];
#pragma unroll
  for (int i = 0; i < 4; ++i)
#pragma unroll
    for (int e = 0; e < 8; ++e) acc[i][e] = 0.f;

  const float*          Af = (const float*)Ap;
  const unsigned short* Ab = (const unsigned short*)Ap;
  const unsigned short* Brow = Bp + (long long)(n0 + lhalf) * K + bk;

  for (int k0 = 0; k0 < K; k0 += 32) {
    V16U a;
    if (AF32) {
      const float* ap = Af + (long long)am * K + k0 + ak;
      float4 f0 = *(const float4*)(ap);
      float4 f1 = *(const float4*)(ap + 4);
      float4 f2 = *(const float4*)(ap + 16);
      float4 f3 = *(const float4*)(ap + 20);
      a.w[0] = f2bf_pk(f0.x, f0.y);
      a.w[1] = f2bf_pk(f0.z, f0.w);
      a.w[2] = f2bf_pk(f1.x, f1.y);
      a.w[3] = f2bf_pk(f1.z, f1.w);
      a.w[4] = f2bf_pk(f2.x, f2.y);
      a.w[5] = f2bf_pk(f2.z, f2.w);
      a.w[6] = f2bf_pk(f3.x, f3.y);
      a.w[7] = f2bf_pk(f3.z, f3.w);
    } else {
      const unsigned short* ap = Ab + (long long)am * K + k0 + ak;
      a.q[0] = *(const uint4*)(ap);
      a.q[1] = *(const uint4*)(ap + 16);
    }
#pragma unroll
    for (int i = 0; i < 4; ++i) {
      const unsigned short* bp = Brow + (long long)(i << 4) * K + k0;
      V16U b;
      b.q[0] = *(const uint4*)(bp);
      b.q[1] = *(const uint4*)(bp + 16);
      acc[i] = __builtin_amdgcn_wmma_f32_16x16x32_bf16(
                 false, a.v, false, b.v, (short)0, acc[i], false, false);
    }
  }

  // ---- epilogue ----
  const int mbase = m0 + (lsel << 3);
  // Row map (independent of n): 2 divisions total, incremental wrap for the
  // remaining 7 rows instead of a div+mod per element.
  int orow[8];
  if (SWAP) {
    const int outer = M / inner;
    int q = mbase / inner;
    int rem = mbase - q * inner;
#pragma unroll
    for (int r = 0; r < 8; ++r) {
      orow[r] = rem * outer + q;
      if (++rem == inner) { rem = 0; ++q; }
    }
  } else {
#pragma unroll
    for (int r = 0; r < 8; ++r) orow[r] = mbase + r;
  }

#pragma unroll
  for (int i = 0; i < 4; ++i) {
    const int n = n0 + (i << 4) + lhalf;
    if (n >= Nreal) continue;
    float bv = 0.f;
    if (HAS_BIAS) bv = bias[n];
#pragma unroll
    for (int r = 0; r < 8; ++r) {
      float v = acc[i][r] + bv;
      if (HAS_CIN) v += Cin[(long long)(mbase + r) * ldout + n];
      if (OUT_BF16) ((unsigned short*)Op)[(long long)orow[r] * ldout + n] = f2bf(v);
      else          ((float*)Op)[(long long)orow[r] * ldout + n] = v;
    }
  }
}

// ---------------------------------------------------------------------------
// LSTM cell elementwise update (gate order i, f, g, o)
// ---------------------------------------------------------------------------
__global__ void lstm_cell(const float* __restrict__ g, float* __restrict__ c,
                          unsigned short* __restrict__ h,
                          unsigned short* __restrict__ h_hist, int hoff)
{
  int idx = blockIdx.x * blockDim.x + threadIdx.x;
  if (idx >= B_SZ * H_DIM) return;
  const int m = idx >> 9, j = idx & 511;
  const float* gr = g + m * G4;
  const float ig = 1.f / (1.f + __expf(-gr[j]));
  const float fg = 1.f / (1.f + __expf(-gr[512 + j]));
  const float gg = tanhf(gr[1024 + j]);
  const float og = 1.f / (1.f + __expf(-gr[1536 + j]));
  const float cn = fg * c[idx] + ig * gg;
  c[idx] = cn;
  const unsigned short hb = f2bf(og * tanhf(cn));
  h[idx] = hb;
  if (h_hist) h_hist[hoff + idx] = hb;
}

// -------------------- prep kernels (all 512-wide rows -> shift/mask) -------
__global__ void k_cvt(const float* __restrict__ s, unsigned short* __restrict__ d, int n) {
  int i = blockIdx.x * blockDim.x + threadIdx.x;
  if (i < n) d[i] = f2bf(s[i]);
}
// copy a 512-wide column slice [c0, c0+512) of an sld-wide matrix -> bf16
__global__ void k_cvt_cols512(const float* __restrict__ s, unsigned short* __restrict__ d,
                              int rows, int sld, int c0) {
  int i = blockIdx.x * blockDim.x + threadIdx.x;
  if (i >= (rows << 9)) return;
  int r = i >> 9, cc = i & 511;
  d[i] = f2bf(s[(long long)r * sld + c0 + cc]);
}
// convert src (srows x 512) into dst (drows x 512), zero-filling padded rows
__global__ void k_cvt_pad512(const float* __restrict__ s, unsigned short* __restrict__ d,
                             int srows, int drows) {
  int i = blockIdx.x * blockDim.x + threadIdx.x;
  if (i >= (drows << 9)) return;
  int r = i >> 9;
  d[i] = (r < srows) ? f2bf(s[i]) : (unsigned short)0;
}
__global__ void k_bias2(const float* a, const float* b, float* d, int n) {
  int i = blockIdx.x * blockDim.x + threadIdx.x;
  if (i < n) d[i] = a[i] + b[i];
}
// decoder word gather -> bf16, (T, B, DW) layout; ids = [<start>, labels[:, :-1]]
__global__ void k_gather(const int* __restrict__ labels, const float* __restrict__ emb,
                         unsigned short* __restrict__ d) {
  int i = blockIdx.x * blockDim.x + threadIdx.x;
  if (i >= T_SEQ * B_SZ * DW_DIM) return;
  int t = i >> 15;                 // / (64*512)
  int r = i & 32767;
  int b = r >> 9, j = r & 511;
  int id = (t == 0) ? START_ID : labels[b * T_SEQ + t - 1];
  d[i] = f2bf(emb[(long long)id * DW_DIM + j]);
}
__global__ void k_zero(unsigned short* h, float* c, int n) {
  int i = blockIdx.x * blockDim.x + threadIdx.x;
  if (i < n) { h[i] = 0; c[i] = 0.f; }
}

// ---------------------------------------------------------------------------
extern "C" void kernel_launch(void* const* d_in, const int* in_sizes, int n_in,
                              void* d_out, int out_size, void* d_ws, size_t ws_size,
                              hipStream_t stream) {
  (void)in_sizes; (void)n_in; (void)out_size; (void)ws_size;
  const float* feats   = (const float*)d_in[0];
  const int*   labels  = (const int*)d_in[1];
  const float* W_frame = (const float*)d_in[2];
  const float* b_frame = (const float*)d_in[3];
  const float* embed_w = (const float*)d_in[4];
  const float* W_ih    = (const float*)d_in[5];
  const float* W_hh    = (const float*)d_in[6];
  const float* b_ih    = (const float*)d_in[7];
  const float* b_hh    = (const float*)d_in[8];
  const float* W_out   = (const float*)d_in[9];
  const float* b_out   = (const float*)d_in[10];
  float* out = (float*)d_out;

  char* ws = (char*)d_ws;
  size_t off = 0;
  auto alloc = [&](size_t bytes) -> void* {
    void* p = ws + off; off = (off + bytes + 255) & ~(size_t)255; return p;
  };
  unsigned short* wihv   = (unsigned short*)alloc((size_t)G4 * E_DIM * 2);
  unsigned short* wihw   = (unsigned short*)alloc((size_t)G4 * DW_DIM * 2);
  unsigned short* whh    = (unsigned short*)alloc((size_t)G4 * H_DIM * 2);
  unsigned short* wout_b = (unsigned short*)alloc((size_t)V_PAD * H_DIM * 2);
  unsigned short* wfrm_b = (unsigned short*)alloc((size_t)E_DIM * F_DIM * 2);
  float*          bcomb  = (float*)alloc((size_t)G4 * 4);
  unsigned short* emb_b  = (unsigned short*)alloc((size_t)T_SEQ * B_SZ * E_DIM * 2);   // (T,B,E)
  unsigned short* demb_b = (unsigned short*)alloc((size_t)T_SEQ * B_SZ * DW_DIM * 2);  // (T,B,DW)
  float*          g_enc  = (float*)alloc((size_t)T_SEQ * B_SZ * G4 * 4);               // (T,B,4H)
  float*          g_dec  = (float*)alloc((size_t)T_SEQ * B_SZ * G4 * 4);
  float*          g_step = (float*)alloc((size_t)B_SZ * G4 * 4);
  unsigned short* h_b    = (unsigned short*)alloc((size_t)B_SZ * H_DIM * 2);
  float*          c_f    = (float*)alloc((size_t)B_SZ * H_DIM * 4);
  unsigned short* h_hist = (unsigned short*)alloc((size_t)T_SEQ * B_SZ * H_DIM * 2);   // (T,B,H)

  auto cdiv = [](long long a, long long b) { return (int)((a + b - 1) / b); };
  auto gemm_blocks = [&](int M, int Npad) {
    long long waves = (long long)(M >> 4) * (Npad >> 6);
    return (int)((waves * 32 + 127) / 128);
  };

  // ---- weight prep ----
  k_cvt_cols512<<<cdiv((long long)G4 * E_DIM, 256), 256, 0, stream>>>(W_ih, wihv, G4, E_DIM + DW_DIM, 0);
  k_cvt_cols512<<<cdiv((long long)G4 * DW_DIM, 256), 256, 0, stream>>>(W_ih, wihw, G4, E_DIM + DW_DIM, E_DIM);
  k_cvt<<<cdiv((long long)G4 * H_DIM, 256), 256, 0, stream>>>(W_hh, whh, G4 * H_DIM);
  k_cvt<<<cdiv((long long)E_DIM * F_DIM, 256), 256, 0, stream>>>(W_frame, wfrm_b, E_DIM * F_DIM);
  k_cvt_pad512<<<cdiv((long long)V_PAD * H_DIM, 256), 256, 0, stream>>>(W_out, wout_b, V_DIM, V_PAD);
  k_bias2<<<cdiv(G4, 256), 256, 0, stream>>>(b_ih, b_hh, bcomb, G4);
  k_gather<<<cdiv((long long)T_SEQ * B_SZ * DW_DIM, 256), 256, 0, stream>>>(labels, embed_w, demb_b);
  k_zero<<<cdiv(B_SZ * H_DIM, 256), 256, 0, stream>>>(h_b, c_f, B_SZ * H_DIM);

  // ---- frame embedding: emb(T,B,E) = feats(B,T,F) @ W_frame^T + b_frame ----
  gemm_wmma<true, false, true, true, true>
      <<<gemm_blocks(B_SZ * T_SEQ, E_DIM), 128, 0, stream>>>(
      feats, wfrm_b, nullptr, b_frame, emb_b,
      B_SZ * T_SEQ, F_DIM, E_DIM, E_DIM, E_DIM, T_SEQ);

  // ---- all input-side gate preactivations (hoisted out of recurrence) ----
  gemm_wmma<false, false, true, false, false>
      <<<gemm_blocks(T_SEQ * B_SZ, G4), 128, 0, stream>>>(
      emb_b, wihv, nullptr, bcomb, g_enc,
      T_SEQ * B_SZ, E_DIM, G4, G4, G4, 1);
  gemm_wmma<false, false, true, false, false>
      <<<gemm_blocks(T_SEQ * B_SZ, G4), 128, 0, stream>>>(
      demb_b, wihw, nullptr, bcomb, g_dec,
      T_SEQ * B_SZ, DW_DIM, G4, G4, G4, 1);

  // ---- sequential recurrence: gates = g_pre[t] + h @ W_hh^T ----
  const int sb = gemm_blocks(B_SZ, G4);
  const int cb = cdiv(B_SZ * H_DIM, 256);
  for (int t = 0; t < T_SEQ; ++t) {  // encoder
    gemm_wmma<false, true, false, false, false><<<sb, 128, 0, stream>>>(
        h_b, whh, g_enc + (size_t)t * B_SZ * G4, nullptr, g_step,
        B_SZ, H_DIM, G4, G4, G4, 1);
    lstm_cell<<<cb, 256, 0, stream>>>(g_step, c_f, h_b, nullptr, 0);
  }
  for (int t = 0; t < T_SEQ; ++t) {  // decoder (records h_t)
    gemm_wmma<false, true, false, false, false><<<sb, 128, 0, stream>>>(
        h_b, whh, g_dec + (size_t)t * B_SZ * G4, nullptr, g_step,
        B_SZ, H_DIM, G4, G4, G4, 1);
    lstm_cell<<<cb, 256, 0, stream>>>(g_step, c_f, h_b, h_hist, t * B_SZ * H_DIM);
  }

  // ---- logits(B,T,V) = h_hist(T,B,H) @ W_out^T + b_out ----
  gemm_wmma<false, false, true, false, true>
      <<<gemm_blocks(T_SEQ * B_SZ, V_PAD), 128, 0, stream>>>(
      h_hist, wout_b, nullptr, b_out, out,
      T_SEQ * B_SZ, H_DIM, V_PAD, V_DIM, V_DIM, B_SZ);
}
